// ScaledDotProductAttention_31799937860179
// MI455X (gfx1250) — matrix operand order repaired and checked by reading, not compiled, not run on hardware
//
#include <hip/hip_runtime.h>

typedef __attribute__((ext_vector_type(16))) _Float16 v16h;
typedef __attribute__((ext_vector_type(8)))  _Float16 v8h;
typedef __attribute__((ext_vector_type(2)))  _Float16 h2f;
typedef __attribute__((ext_vector_type(8)))  float    v8f;

namespace {
constexpr int kB = 8, kH = 16, kS = 1024, kD = 64;
constexpr int QT = 16;   // query rows per wave
constexpr int NW = 4;    // waves per block
constexpr int KT = 32;   // keys per main-loop iteration
// 1/sqrt(64) * log2(e): scores produced directly in log2 domain -> v_exp_f32
constexpr float kQScale = 0.125f * 1.4426950408889634f;
}

union V16 { v16h v; v8h h[2]; };

// Bare v_max for values that are NOT direct WMMA results (no canonicalize,
// and inline asm is invisible to the WMMA->VALU hazard recognizer, so its
// inputs must already be ALU-produced).
__device__ __forceinline__ float fmaxq(float a, float b) {
  float r;
  asm("v_max_num_f32 %0, %1, %2" : "=v"(r) : "v"(a), "v"(b));
  return r;
}

// 16-lane-half butterfly reductions with the DPP permute FUSED into the
// max/add (one instruction per step). Masks {1,2,7,15} (quad_perm xor1/xor2,
// row_half_mirror, row_mirror) are GF(2)-independent -> full 16-lane reduce.
// Inputs must be ALU results (see hazard note above).
__device__ __forceinline__ float half16_max(float x) {
  asm("v_max_num_f32_dpp %0, %1, %1 quad_perm:[1,0,3,2] row_mask:0xf bank_mask:0xf bound_ctrl:1"
      : "=v"(x) : "v"(x));
  asm("v_max_num_f32_dpp %0, %1, %1 quad_perm:[2,3,0,1] row_mask:0xf bank_mask:0xf bound_ctrl:1"
      : "=v"(x) : "v"(x));
  asm("v_max_num_f32_dpp %0, %1, %1 row_half_mirror row_mask:0xf bank_mask:0xf bound_ctrl:1"
      : "=v"(x) : "v"(x));
  asm("v_max_num_f32_dpp %0, %1, %1 row_mirror row_mask:0xf bank_mask:0xf bound_ctrl:1"
      : "=v"(x) : "v"(x));
  return x;
}
__device__ __forceinline__ float half16_sum(float x) {
  asm("v_add_f32_dpp %0, %1, %1 quad_perm:[1,0,3,2] row_mask:0xf bank_mask:0xf bound_ctrl:1"
      : "=v"(x) : "v"(x));
  asm("v_add_f32_dpp %0, %1, %1 quad_perm:[2,3,0,1] row_mask:0xf bank_mask:0xf bound_ctrl:1"
      : "=v"(x) : "v"(x));
  asm("v_add_f32_dpp %0, %1, %1 row_half_mirror row_mask:0xf bank_mask:0xf bound_ctrl:1"
      : "=v"(x) : "v"(x));
  asm("v_add_f32_dpp %0, %1, %1 row_mirror row_mask:0xf bank_mask:0xf bound_ctrl:1"
      : "=v"(x) : "v"(x));
  return x;
}

__global__ __launch_bounds__(NW * 32)
void fa_fwd_wmma_f16(const float* __restrict__ Qg, const float* __restrict__ Kg,
                     const float* __restrict__ Vg, float* __restrict__ Og) {
  // Double-buffered, block-shared f16 staging tiles. Row pitches are 16B
  // multiples (72*2=144B, 40*2=80B) so fragment loads are aligned b128 DS ops.
  __shared__ __align__(16) _Float16 ldsK[2][KT][kD + 8];   // [key][d]
  __shared__ __align__(16) _Float16 ldsV[2][kD][KT + 8];   // [d][key] (transposed)
  __shared__ __align__(16) _Float16 pbuf[NW][QT][KT + 8];  // per-wave P transpose

  const int lane = threadIdx.x & 31;
  const int wv   = threadIdx.x >> 5;
  const int hf   = lane >> 4;   // 16-lane half
  const int hl   = lane & 15;

  const int bh = blockIdx.x >> 4;
  const int q0 = (blockIdx.x & 15) * (QT * NW) + wv * QT;

  const float* Qb = Qg + (size_t)bh * kS * kD;
  const float* Kb = Kg + (size_t)bh * kS * kD;
  const float* Vb = Vg + (size_t)bh * kS * kD;
  float*       Ob = Og + (size_t)bh * kS * kD;

  // ---- Q tile (16x64) -> two f16 A fragments (d chunks 0..31 / 32..63).
  // A layout: lane (hf,hl) = row hl; halfs 0..15 = cols hf*16..+16
  // (within each 32-wide chunk).
  v16h aQ0, aQ1;
  {
    const float* qrow = Qb + (size_t)(q0 + hl) * kD;
#pragma unroll
    for (int c = 0; c < 2; ++c) {
      const float4 f0 = *(const float4*)(qrow + c * 32 + hf * 16);
      const float4 f1 = *(const float4*)(qrow + c * 32 + hf * 16 + 4);
      const float4 f2 = *(const float4*)(qrow + c * 32 + hf * 16 + 8);
      const float4 f3 = *(const float4*)(qrow + c * 32 + hf * 16 + 12);
      v16h a;
      a[0] = (_Float16)(f0.x * kQScale); a[1] = (_Float16)(f0.y * kQScale);
      a[2] = (_Float16)(f0.z * kQScale); a[3] = (_Float16)(f0.w * kQScale);
      a[4] = (_Float16)(f1.x * kQScale); a[5] = (_Float16)(f1.y * kQScale);
      a[6] = (_Float16)(f1.z * kQScale); a[7] = (_Float16)(f1.w * kQScale);
      a[8]  = (_Float16)(f2.x * kQScale); a[9]  = (_Float16)(f2.y * kQScale);
      a[10] = (_Float16)(f2.z * kQScale); a[11] = (_Float16)(f2.w * kQScale);
      a[12] = (_Float16)(f3.x * kQScale); a[13] = (_Float16)(f3.y * kQScale);
      a[14] = (_Float16)(f3.z * kQScale); a[15] = (_Float16)(f3.w * kQScale);
      if (c == 0) aQ0 = a; else aQ1 = a;
    }
  }

  // ---- Cooperative staging: wave wv converts K keys [wv*8,wv*8+8) row-major
  // and V dims [wv*16,wv*16+16) transposed. Load->regs first, store later.
  const int krow = wv * 8 + (lane >> 2);
  const int kcol = (lane & 3) * 16;
  const int vk0  = 2 * hl;              // even key
  const int vdb  = wv * 16 + 8 * hf;    // d base

  float4 kf0, kf1, kf2, kf3, va0, va1, vb0, vb1;
  auto stage_load = [&](int kt2) {
    const float* ks = Kb + (size_t)(kt2 + krow) * kD + kcol;
    kf0 = ((const float4*)ks)[0]; kf1 = ((const float4*)ks)[1];
    kf2 = ((const float4*)ks)[2]; kf3 = ((const float4*)ks)[3];
    const float* vs = Vb + (size_t)(kt2 + vk0) * kD + vdb;
    va0 = ((const float4*)vs)[0]; va1 = ((const float4*)vs)[1];
    const float* vt = vs + kD;
    vb0 = ((const float4*)vt)[0]; vb1 = ((const float4*)vt)[1];
  };
  auto stage_store = [&](int bufi) {
    v8h h0, h1;
    h0[0] = (_Float16)kf0.x; h0[1] = (_Float16)kf0.y;
    h0[2] = (_Float16)kf0.z; h0[3] = (_Float16)kf0.w;
    h0[4] = (_Float16)kf1.x; h0[5] = (_Float16)kf1.y;
    h0[6] = (_Float16)kf1.z; h0[7] = (_Float16)kf1.w;
    h1[0] = (_Float16)kf2.x; h1[1] = (_Float16)kf2.y;
    h1[2] = (_Float16)kf2.z; h1[3] = (_Float16)kf2.w;
    h1[4] = (_Float16)kf3.x; h1[5] = (_Float16)kf3.y;
    h1[6] = (_Float16)kf3.z; h1[7] = (_Float16)kf3.w;
    *(v8h*)&ldsK[bufi][krow][kcol]     = h0;
    *(v8h*)&ldsK[bufi][krow][kcol + 8] = h1;
    const float av[8] = {va0.x, va0.y, va0.z, va0.w, va1.x, va1.y, va1.z, va1.w};
    const float bv[8] = {vb0.x, vb0.y, vb0.z, vb0.w, vb1.x, vb1.y, vb1.z, vb1.w};
#pragma unroll
    for (int i = 0; i < 8; ++i) {
      h2f pk;
      pk[0] = (_Float16)av[i];          // key vk0
      pk[1] = (_Float16)bv[i];          // key vk0+1
      *(h2f*)&ldsV[bufi][vdb + i][vk0] = pk;  // b32 DS store
    }
  };

  v8f o0 = {}, o1 = {}, o2 = {}, o3 = {};
  float mrow[8], lrow[8];   // lrow is LANE-PARTIAL; reduced once in epilogue
#pragma unroll
  for (int r = 0; r < 8; ++r) { mrow[r] = -1e30f; lrow[r] = 0.0f; }

  stage_load(0);
  stage_store(0);
  __syncthreads();

  for (int kt = 0; kt < kS; kt += KT) {
    const int cur = (kt >> 5) & 1;

    // Start next tile's global loads; they retire during this tile's compute.
    if (kt + KT < kS) stage_load(kt + KT);
    if (kt + 2 * KT < kS) {
      __builtin_prefetch(Kb + (size_t)(kt + 2 * KT) * kD + lane * 64, 0, 3);
      __builtin_prefetch(Vb + (size_t)(kt + 2 * KT) * kD + lane * 64, 0, 3);
    }

    // ---- All four K fragments first (8 x ds_load_b128), then four WMMAs.
    // Interleaved s0/s1 accumulator chains keep back-to-back WMMAs independent.
    V16 k00, k01, k10, k11;
    {
      const _Float16* kp0 = &ldsK[cur][hl][hf * 16];
      const _Float16* kp1 = &ldsK[cur][16 + hl][hf * 16];
      k00.h[0] = *(const v8h*)kp0;        k00.h[1] = *(const v8h*)(kp0 + 8);
      k01.h[0] = *(const v8h*)(kp0 + 32); k01.h[1] = *(const v8h*)(kp0 + 40);
      k10.h[0] = *(const v8h*)kp1;        k10.h[1] = *(const v8h*)(kp1 + 8);
      k11.h[0] = *(const v8h*)(kp1 + 32); k11.h[1] = *(const v8h*)(kp1 + 40);
    }
    v8f s0 = {}, s1 = {};
    s0 = __builtin_amdgcn_wmma_f32_16x16x32_f16(false, aQ0, false, k00.v,
                                                (short)0, s0, false, false);
    s1 = __builtin_amdgcn_wmma_f32_16x16x32_f16(false, aQ0, false, k10.v,
                                                (short)0, s1, false, false);
    s0 = __builtin_amdgcn_wmma_f32_16x16x32_f16(false, aQ1, false, k01.v,
                                                (short)0, s0, false, false);
    s1 = __builtin_amdgcn_wmma_f32_16x16x32_f16(false, aQ1, false, k11.v,
                                                (short)0, s1, false, false);

    // ---- Online softmax. The pairwise max is a COMPILER op (fmaxf) so the
    // WMMA->VALU RAW hazard pass sees the first read of the WMMA results and
    // inserts the required NOPs; the asm butterfly then only touches ALU
    // results. Row stats live per 16-lane half.
    float p0[8], p1[8];
#pragma unroll
    for (int r = 0; r < 8; ++r) {
      const float tmax = half16_max(fmaxf(s0[r], s1[r]));
      const float mnew = fmaxq(mrow[r], tmax);
      const float corr = __builtin_amdgcn_exp2f(mrow[r] - mnew);
      mrow[r] = mnew;
      p0[r] = __builtin_amdgcn_exp2f(s0[r] - mnew);
      p1[r] = __builtin_amdgcn_exp2f(s1[r] - mnew);
      lrow[r] = lrow[r] * corr + (p0[r] + p1[r]);
      o0[r] *= corr; o1[r] *= corr; o2[r] *= corr; o3[r] *= corr;
    }

    // ---- P (16x32) C-layout -> per-wave LDS row-major f16 (DS is in-order
    // per wave; the compiler inserts the minimal dscnt wait for the RAW).
#pragma unroll
    for (int r = 0; r < 8; ++r) {
      pbuf[wv][r + 8 * hf][hl]      = (_Float16)p0[r];
      pbuf[wv][r + 8 * hf][16 + hl] = (_Float16)p1[r];
    }
    v16h aP;
    {
      const _Float16* pp = &pbuf[wv][hl][0];
      V16 u;
      u.h[0] = *(const v8h*)(pp + hf * 16);
      u.h[1] = *(const v8h*)(pp + hf * 16 + 8);
      aP = u.v;
    }

    // ---- All four V fragments, then four independent-accumulator WMMAs.
    V16 vf0, vf1, vf2, vf3;
    {
      const _Float16* vp0 = &ldsV[cur][ 0 + hl][hf * 16];
      const _Float16* vp1 = &ldsV[cur][16 + hl][hf * 16];
      const _Float16* vp2 = &ldsV[cur][32 + hl][hf * 16];
      const _Float16* vp3 = &ldsV[cur][48 + hl][hf * 16];
      vf0.h[0] = *(const v8h*)vp0; vf0.h[1] = *(const v8h*)(vp0 + 8);
      vf1.h[0] = *(const v8h*)vp1; vf1.h[1] = *(const v8h*)(vp1 + 8);
      vf2.h[0] = *(const v8h*)vp2; vf2.h[1] = *(const v8h*)(vp2 + 8);
      vf3.h[0] = *(const v8h*)vp3; vf3.h[1] = *(const v8h*)(vp3 + 8);
    }
    o0 = __builtin_amdgcn_wmma_f32_16x16x32_f16(false, aP, false, vf0.v,
                                                (short)0, o0, false, false);
    o1 = __builtin_amdgcn_wmma_f32_16x16x32_f16(false, aP, false, vf1.v,
                                                (short)0, o1, false, false);
    o2 = __builtin_amdgcn_wmma_f32_16x16x32_f16(false, aP, false, vf2.v,
                                                (short)0, o2, false, false);
    o3 = __builtin_amdgcn_wmma_f32_16x16x32_f16(false, aP, false, vf3.v,
                                                (short)0, o3, false, false);

    // Convert + publish next tile, then one barrier covers "all reads of cur
    // done" and "all writes of cur^1 done" (double buffering).
    if (kt + KT < kS) stage_store(cur ^ 1);
    __syncthreads();
  }

  // ---- Epilogue: reduce the lane-partial denominators once, normalize, store.
  // (o reads and the inv multiply are compiler ops -> WMMA hazard covered.)
#pragma unroll
  for (int r = 0; r < 8; ++r) {
    const float inv = __builtin_amdgcn_rcpf(half16_sum(lrow[r]));
    const size_t row = (size_t)(q0 + r + 8 * hf) * kD;
    Ob[row +  0 + hl] = o0[r] * inv;
    Ob[row + 16 + hl] = o1[r] * inv;
    Ob[row + 32 + hl] = o2[r] * inv;
    Ob[row + 48 + hl] = o3[r] * inv;
  }
}

extern "C" void kernel_launch(void* const* d_in, const int* in_sizes, int n_in,
                              void* d_out, int out_size, void* d_ws, size_t ws_size,
                              hipStream_t stream) {
  (void)in_sizes; (void)n_in; (void)out_size; (void)d_ws; (void)ws_size;
  const float* Q = (const float*)d_in[0];
  const float* K = (const float*)d_in[1];
  const float* V = (const float*)d_in[2];
  float* O = (float*)d_out;
  const int qblocks = kS / (QT * NW);   // 16
  dim3 grid(kB * kH * qblocks);         // 2048 blocks
  dim3 block(NW * 32);                  // 4 wave32 waves
  fa_fwd_wmma_f16<<<grid, block, 0, stream>>>(Q, K, V, O);
}
